// MatryoshkaTranscoder_21303037788824
// MI455X (gfx1250) — matrix operand order repaired and checked by reading, not compiled, not run on hardware
//
#include <hip/hip_runtime.h>

typedef __attribute__((ext_vector_type(16))) __bf16 v16bf;
typedef __attribute__((ext_vector_type(8)))  float  v8f;

#define BATCH   4096
#define IN_DIM  2048
#define OUT_DIM 2048
#define LATENT  32768

#define BM 128
#define BN 128
#define BK 32
#define LDSW 40   // ushorts per LDS row: 32 data + 8 pad (80B stride keeps 16B-aligned frags)

// pack two fp32 -> packed bf16x2 (round-half-up) with a single v_perm_b32
__device__ __forceinline__ unsigned pack_bf16x2(float lo, float hi) {
    return __builtin_amdgcn_perm(__float_as_uint(hi) + 0x8000u,
                                 __float_as_uint(lo) + 0x8000u,
                                 0x07060302u);
}

// C(MxN) = A(MxK) @ B(NxK)^T [+ bias, optional jump_relu], A/B fp32 row-major (K contiguous)
template<int EPILOGUE>  // 0: bias only (decoder)   1: bias + jump_relu (encoder)
__global__ __launch_bounds__(256)
void gemm_bf16(const float* __restrict__ A, const float* __restrict__ Bm,
               const float* __restrict__ bias, float* __restrict__ C,
               int M, int N, int K)
{
    __shared__ unsigned short As[2][BM * LDSW];
    __shared__ unsigned short Bs[2][BN * LDSW];

    const int tid  = threadIdx.x;
    const int m0   = blockIdx.y * BM;
    const int n0   = blockIdx.x * BN;

    const int wave = tid >> 5;
    const int lane = tid & 31;
    const int g    = lane >> 4;          // half-wave group
    const int mloc = lane & 15;
    const int wm   = (wave & 3) * 32;    // wave M offset: 4 waves x 32 rows
    const int wn   = (wave >> 2) * 64;   // wave N offset: 2 waves x 64 cols

    v8f acc[2][4];
    #pragma unroll
    for (int s = 0; s < 2; ++s)
        #pragma unroll
        for (int f = 0; f < 4; ++f)
            acc[s][f] = (v8f){0.f,0.f,0.f,0.f,0.f,0.f,0.f,0.f};

    // staging registers: 128x32 fp32 tile each for A and B = 1024 float4 = 4/thread each
    float4 ra[4], rb[4];

    auto load_tile = [&](int t) {
        const int kk = t * BK;
        #pragma unroll
        for (int j = 0; j < 4; ++j) {
            const int l = tid + j * 256;
            const int r = l >> 3, c4 = l & 7;
            ra[j] = *(const float4*)(A  + (size_t)(m0 + r) * K + kk + c4 * 4);
            rb[j] = *(const float4*)(Bm + (size_t)(n0 + r) * K + kk + c4 * 4);
        }
    };
    auto store_tile = [&](int buf) {
        #pragma unroll
        for (int j = 0; j < 4; ++j) {
            const int l = tid + j * 256;
            const int r = l >> 3, c4 = l & 7;
            *(uint2*)&As[buf][r * LDSW + c4 * 4] =
                make_uint2(pack_bf16x2(ra[j].x, ra[j].y), pack_bf16x2(ra[j].z, ra[j].w));
            *(uint2*)&Bs[buf][r * LDSW + c4 * 4] =
                make_uint2(pack_bf16x2(rb[j].x, rb[j].y), pack_bf16x2(rb[j].z, rb[j].w));
        }
    };

    const int ntiles = K / BK;

    load_tile(0);
    store_tile(0);
    __syncthreads();

    for (int t = 0; t < ntiles; ++t) {
        const int cur = t & 1;
        if (t + 1 < ntiles) load_tile(t + 1);   // global loads overlap the WMMA cluster below

        // ---- load ALL fragments first (single dscnt wait), then back-to-back WMMAs
        union Frag { int4 q[2]; v16bf v; };
        Frag af[2], bfr[4];
        #pragma unroll
        for (int s = 0; s < 2; ++s) {
            const int rbase = (wm + s * 16 + mloc) * LDSW;
            af[s].q[0] = *(const int4*)&As[cur][rbase + g * 8];
            af[s].q[1] = *(const int4*)&As[cur][rbase + 16 + g * 8];
        }
        #pragma unroll
        for (int f = 0; f < 4; ++f) {
            const int rbase = (wn + f * 16 + mloc) * LDSW;
            bfr[f].q[0] = *(const int4*)&Bs[cur][rbase + g * 8];
            bfr[f].q[1] = *(const int4*)&Bs[cur][rbase + 16 + g * 8];
        }
        #pragma unroll
        for (int s = 0; s < 2; ++s)
            #pragma unroll
            for (int f = 0; f < 4; ++f)
                acc[s][f] = __builtin_amdgcn_wmma_f32_16x16x32_bf16(
                    false, af[s].v, false, bfr[f].v, (short)0, acc[s][f], false, false);

        if (t + 1 < ntiles) store_tile(cur ^ 1);  // convert + stage next tile into other buffer
        __syncthreads();
    }

    // ---- epilogue: D layout VGPR j -> (M = j + 8*g, N = lane%16)
    #pragma unroll
    for (int f = 0; f < 4; ++f) {
        const int n  = n0 + wn + f * 16 + mloc;
        const float bv = bias[n];
        #pragma unroll
        for (int s = 0; s < 2; ++s) {
            #pragma unroll
            for (int j = 0; j < 8; ++j) {
                const int m = m0 + wm + s * 16 + g * 8 + j;
                float val = acc[s][f][j] + bv;
                if (EPILOGUE == 1) // jump_relu: relu(x) + 1.0*(x > 1.0)
                    val = fmaxf(val, 0.f) + ((val > 1.0f) ? 1.0f : 0.f);
                C[(size_t)m * N + n] = val;
            }
        }
    }
}

// In-place nested per-segment top-k (by |value|) masking of z, one block per (segment, row).
// Deterministic: integer radix binary-search on abs bit patterns + index-ordered tie-break.
__global__ __launch_bounds__(256)
void nested_topk(float* __restrict__ z)
{
    const int starts[5] = {0, 2048, 4096, 8192, 16384};
    const int lens[5]   = {2048, 2048, 4096, 8192, 16384};
    const int ks[5]     = {8, 16, 32, 64, 128};

    const int seg = blockIdx.x;
    const int row = blockIdx.y;
    const int len = lens[seg];
    const int k   = ks[seg];
    float* zrow = z + (size_t)row * LATENT + starts[seg];

    extern __shared__ float buf[];   // up to 16384 floats (64 KB of 320 KB WGP LDS)
    __shared__ int scnt;

    const int tid = threadIdx.x;
    for (int i = tid * 4; i < len; i += 256 * 4)
        *(float4*)&buf[i] = *(const float4*)&zrow[i];
    __syncthreads();

    // largest integer threshold t with count(|x| bits >= t) >= k
    unsigned t = 0;
    for (int bit = 30; bit >= 0; --bit) {
        const unsigned cand = t | (1u << bit);
        if (tid == 0) scnt = 0;
        __syncthreads();
        int c = 0;
        for (int i = tid; i < len; i += 256) {
            const unsigned key = __float_as_uint(buf[i]) & 0x7FFFFFFFu;
            c += (key >= cand) ? 1 : 0;
        }
        atomicAdd(&scnt, c);
        __syncthreads();
        if (scnt >= k) t = cand;
        __syncthreads();
    }

    // strictly-greater count (all of these are kept)
    if (tid == 0) scnt = 0;
    __syncthreads();
    {
        int c = 0;
        for (int i = tid; i < len; i += 256) {
            const unsigned key = __float_as_uint(buf[i]) & 0x7FFFFFFFu;
            c += (key > t) ? 1 : 0;
        }
        atomicAdd(&scnt, c);
    }
    __syncthreads();
    const int c_gt = scnt;

    // masked write-back; ties at t kept in index order (deterministic)
    for (int i = tid; i < len; i += 256) {
        const float v = buf[i];
        const unsigned key = __float_as_uint(v) & 0x7FFFFFFFu;
        float outv = 0.f;
        if (key > t) {
            outv = v;
        } else if (key == t) {
            int r = 0;
            for (int j = 0; j < i; ++j)
                r += ((__float_as_uint(buf[j]) & 0x7FFFFFFFu) == t) ? 1 : 0;
            if (c_gt + r < k) outv = v;
        }
        zrow[i] = outv;
    }
}

extern "C" void kernel_launch(void* const* d_in, const int* in_sizes, int n_in,
                              void* d_out, int out_size, void* d_ws, size_t ws_size,
                              hipStream_t stream)
{
    const float* h2    = (const float*)d_in[0];   // (4096, 2048)
    const float* Wenc  = (const float*)d_in[1];   // (32768, 2048)
    const float* benc  = (const float*)d_in[2];   // (32768,)
    const float* Wdec4 = (const float*)d_in[11];  // (2048, 32768)
    const float* bdec4 = (const float*)d_in[12];  // (2048,)

    float* out = (float*)d_out;                    // h_1_recon: (4096, 2048)
    float* z   = out + (size_t)BATCH * OUT_DIM;    // z_sparse region: (4096, 32768)

    dim3 blk(256);

    // 1) encoder GEMM + fused bias + jump_relu -> dense z (written into z_sparse slot)
    gemm_bf16<1><<<dim3(LATENT / BN, BATCH / BM), blk, 0, stream>>>(
        h2, Wenc, benc, z, BATCH, LATENT, IN_DIM);

    // 2) nested top-k masking in place (5 segments x 4096 rows)
    nested_topk<<<dim3(5, BATCH), blk, 16384 * sizeof(float), stream>>>(z);

    // 3) decoder GEMM + fused bias -> h_1_recon
    gemm_bf16<0><<<dim3(OUT_DIM / BN, BATCH / BM), blk, 0, stream>>>(
        z, Wdec4, bdec4, out, BATCH, OUT_DIM, LATENT);
}
